// init_5952824672700
// MI455X (gfx1250) — matrix operand order repaired and checked by reading, not compiled, not run on hardware
//
#include <hip/hip_runtime.h>
#include <hip/hip_bf16.h>

typedef __attribute__((ext_vector_type(16))) __bf16          v16bf;
typedef __attribute__((ext_vector_type(2)))  __bf16          v2bf;
typedef __attribute__((ext_vector_type(8)))  float           v8f;
typedef __attribute__((ext_vector_type(8)))  unsigned short  u16x8;

#define HIDDEN    128
#define NRAD      6
#define EPB       128      // edges per block (8 waves x 16 rows)
#define THREADS   256
#define KSTEPS    13       // 12 for cat @ w_lin, 1 padded block for rbf @ w_rbf1
#define A_STRIDE  424      // 416 + 8 pad bf16 elems; 848B rows keep 16B align, spread banks
#define NFRAG     (KSTEPS * 8 * 512)   // 53248 bf16 of packed B fragments

// LDS layout (bytes):
//   A tile   : [0,      108544)  128 x 424 bf16  (h[i] | h[j] | rbf0 | rbf-pad)
//   B frags  : [108544, 215040)  w_lin (+w_rbf1 block) in WMMA B-fragment order
//   rbf tile : [215040, 219136)  128 x 8 f32
//   WS       : [219136, 223232)  w_rbf0[768] b_rbf0[128] b_lin[128] f32
#define SMEM_BYTES 223232

// ---------------- bf16 conversion ----------------
__device__ __forceinline__ unsigned short f2bf(float f) {
  union { float f; unsigned u; } c; c.f = f;
  unsigned u = c.u + 0x7FFFu + ((c.u >> 16) & 1u);   // round-to-nearest-even
  return (unsigned short)(u >> 16);
}
#if __has_builtin(__builtin_amdgcn_cvt_pk_bf16_f32)
__device__ __forceinline__ unsigned pk2bf(float a, float b) {
  union { v2bf v; unsigned u; } c;
  c.v = __builtin_amdgcn_cvt_pk_bf16_f32(a, b);      // single v_cvt_pk_bf16_f32
  return c.u;
}
#else
__device__ __forceinline__ unsigned pk2bf(float a, float b) {
  return (unsigned)f2bf(a) | ((unsigned)f2bf(b) << 16);
}
#endif

__device__ __forceinline__ float swishf(float x) {
  return x / (1.0f + __expf(-x));                    // x * sigmoid(x)
}

// ---------------- async global -> LDS copies (CDNA5 ASYNCcnt path) ----------------
// Prototypes (from hipcc diagnostics): b128 takes (int4 AS1*, int4 AS3*, imm, imm),
// b32 takes (int AS1*, int AS3*, imm, imm); global source first, non-const pointees.
typedef int v4i __attribute__((vector_size(4 * sizeof(int))));
typedef __attribute__((address_space(1))) v4i  g_v4i;
typedef __attribute__((address_space(3))) v4i  l_v4i;
typedef __attribute__((address_space(1))) int  g_i32;
typedef __attribute__((address_space(3))) int  l_i32;

__device__ __forceinline__ void cp_b128(const void* g, void* l) {
#if __has_builtin(__builtin_amdgcn_global_load_async_to_lds_b128)
  __builtin_amdgcn_global_load_async_to_lds_b128(
      (g_v4i*)(unsigned long long)g,
      (l_v4i*)(unsigned)(unsigned long long)l, 0, 0);
#else
  *(uint4*)l = *(const uint4*)g;
#endif
}
__device__ __forceinline__ void cp_b32(const void* g, void* l) {
#if __has_builtin(__builtin_amdgcn_global_load_async_to_lds_b32)
  __builtin_amdgcn_global_load_async_to_lds_b32(
      (g_i32*)(unsigned long long)g,
      (l_i32*)(unsigned)(unsigned long long)l, 0, 0);
#else
  *(unsigned*)l = *(const unsigned*)g;
#endif
}
__device__ __forceinline__ void async_fence() {
#if __has_builtin(__builtin_amdgcn_global_load_async_to_lds_b128)
#if __has_builtin(__builtin_amdgcn_s_wait_asynccnt)
  __builtin_amdgcn_s_wait_asynccnt(0);
#else
  asm volatile("s_wait_asynccnt 0x0" ::: "memory");
#endif
#endif
}

// Pack B fragments: ks 0..11 from w_lin [384x128]; ks 12 from w_rbf1 [6x128] zero-padded.
// flat = ((ks*8 + nt)*32 + lane)*16 + s ;  k_local = (s>>3)*16 + (lane>>4)*8 + (s&7) ; n = nt*16 + (lane&15)
__global__ void pack_w(const float* __restrict__ w_lin,
                       const float* __restrict__ w_rbf1,
                       unsigned short* __restrict__ wfrag) {
  int idx = blockIdx.x * blockDim.x + threadIdx.x;
  if (idx >= NFRAG) return;
  int s    = idx & 15;
  int lane = (idx >> 4) & 31;
  int tile = idx >> 9;
  int nt = tile & 7, ks = tile >> 3;
  int kl = ((s >> 3) << 4) + ((lane >> 4) << 3) + (s & 7);        // 0..31 within the K-block
  int n  = nt * 16 + (lane & 15);
  float v;
  if (ks < 12)        v = w_lin[(ks * 32 + kl) * HIDDEN + n];
  else if (kl < NRAD) v = w_rbf1[kl * HIDDEN + n];
  else                v = 0.0f;
  wfrag[idx] = f2bf(v);
}

// Per-node bf16 embedding rows: hnode[node] = bf16(emb_table[x[node]]).
__global__ void pack_h(const int* __restrict__ x, const float* __restrict__ emb,
                       unsigned short* __restrict__ hnode, int N) {
  int idx = blockIdx.x * blockDim.x + threadIdx.x;   // one per 4 columns
  if (idx >= N * (HIDDEN / 4)) return;
  int node = idx >> 5, c = (idx & 31) << 2;
  int atom = x[node];
  float4 v = *(const float4*)(emb + (long)atom * HIDDEN + c);
  unsigned* dst = (unsigned*)(hnode + (long)node * HIDDEN + c);
  dst[0] = pk2bf(v.x, v.y);
  dst[1] = pk2bf(v.z, v.w);
}

__global__ __launch_bounds__(THREADS)
void edge_mlp(const float* __restrict__ rbf,
              const int*   __restrict__ eidx_i, const int* __restrict__ eidx_j,
              const unsigned short* __restrict__ hnode,
              const float* __restrict__ w_rbf0, const float* __restrict__ b_rbf0,
              const float* __restrict__ b_lin,
              const unsigned short* __restrict__ wfrag,
              float* __restrict__ out_e1, float* __restrict__ out_e2, int E) {
  extern __shared__ char smem[];
  unsigned short* As   = (unsigned short*)smem;
  unsigned short* Bs   = (unsigned short*)(smem + 108544);
  float*          rbfT = (float*)(smem + 215040);
  float*          WS   = (float*)(smem + 219136);

  const int  t    = threadIdx.x;
  const long base = (long)blockIdx.x * EPB;

  // --- B fragments: 106496 B = 6656 b128 async transfers ---
#pragma unroll
  for (int q = 0; q < 26; ++q) {
    int idx = t + q * THREADS;
    cp_b128(wfrag + idx * 8, Bs + idx * 8);
  }
  // --- rbf tile [128][6] -> [128][8] f32 ---
#pragma unroll
  for (int q = 0; q < 3; ++q) {
    int f = t + q * THREADS;
    if (f < EPB * NRAD) {
      int  e = f / NRAD, r = f - e * NRAD;
      long eg = base + e; if (eg >= E) eg = E - 1;
      cp_b32(rbf + eg * NRAD + r, rbfT + e * 8 + r);
    }
  }
  // --- small weights: w_rbf0 | b_rbf0 | b_lin ---
#pragma unroll
  for (int q = 0; q < 4; ++q) {
    int f = t + q * THREADS;
    if (f < 1024) {
      const float* src;
      if (f < 768)      src = w_rbf0 + f;
      else if (f < 896) src = b_rbf0 + (f - 768);
      else              src = b_lin + (f - 896);
      cp_b32(src, WS + f);
    }
  }
  // --- gather bf16 node rows (single indirection): 512 half-row copy tasks ---
#pragma unroll
  for (int q = 0; q < 2; ++q) {
    int task  = t + q * THREADS;
    int e     = task >> 2;
    int which = (task >> 1) & 1;
    int half  = task & 1;
    long eg = base + e; if (eg >= E) eg = E - 1;
    int node = which ? eidx_j[eg] : eidx_i[eg];
    const unsigned short* src = hnode + (long)node * HIDDEN + half * 64;
    unsigned short*       dst = As + e * A_STRIDE + which * HIDDEN + half * 64;
#pragma unroll
    for (int c = 0; c < 8; ++c) cp_b128(src + c * 8, dst + c * 8);
  }
  async_fence();
  __syncthreads();

  // --- rbf0 = swish(rbf @ w_rbf0 + b_rbf0) -> A cols 256..383 (bf16) ---
  {
    int n     = (t & 63) * 2;       // pair of adjacent columns
    int ebase = (t >> 6) * 32;      // 32 edges per thread
    float w0a[NRAD], w0b[NRAD];
#pragma unroll
    for (int r = 0; r < NRAD; ++r) { w0a[r] = WS[r * HIDDEN + n]; w0b[r] = WS[r * HIDDEN + n + 1]; }
    float b0a = WS[768 + n], b0b = WS[768 + n + 1];
    for (int q = 0; q < 32; ++q) {
      int e  = ebase + q;
      float4 rv0 = *(const float4*)(rbfT + e * 8);
      float4 rv1 = *(const float4*)(rbfT + e * 8 + 4);
      float a0 = b0a + rv0.x * w0a[0] + rv0.y * w0a[1] + rv0.z * w0a[2]
                     + rv0.w * w0a[3] + rv1.x * w0a[4] + rv1.y * w0a[5];
      float a1 = b0b + rv0.x * w0b[0] + rv0.y * w0b[1] + rv0.z * w0b[2]
                     + rv0.w * w0b[3] + rv1.x * w0b[4] + rv1.y * w0b[5];
      *(unsigned*)(As + e * A_STRIDE + 256 + n) = pk2bf(swishf(a0), swishf(a1));
    }
  }
  // --- rbf pad block -> A cols 384..415 (6 bf16 + zeros) ---
  if (t < EPB) {
    int e = t;
    uint4* dst = (uint4*)(As + e * A_STRIDE + 384);
    uint4 v0;
    v0.x = pk2bf(rbfT[e * 8 + 0], rbfT[e * 8 + 1]);
    v0.y = pk2bf(rbfT[e * 8 + 2], rbfT[e * 8 + 3]);
    v0.z = pk2bf(rbfT[e * 8 + 4], rbfT[e * 8 + 5]);
    v0.w = 0u;
    uint4 z; z.x = z.y = z.z = z.w = 0u;
    dst[0] = v0; dst[1] = z; dst[2] = z; dst[3] = z;
  }
  __syncthreads();

  // --- WMMA GEMM: each wave = 16 edges x 128 cols ---
  const int wave = t >> 5, lane = t & 31;
  const int m = lane & 15, halfsel = lane >> 4;
  v8f acc[8], acc2[8];
  v8f zero = {0.f, 0.f, 0.f, 0.f, 0.f, 0.f, 0.f, 0.f};
#pragma unroll
  for (int i = 0; i < 8; ++i) acc[i] = zero;

  // A fragment (ISA 7.12.2): lane(m,half) reads K = ks*32 + g*16 + half*8 + [0..7]
  const unsigned short* Arow = As + (wave * 16 + m) * A_STRIDE + halfsel * 8;
  union AF { v16bf v; u16x8 h[2]; };
#pragma unroll
  for (int ks = 0; ks < 12; ++ks) {
    AF af;
    af.h[0] = *(const u16x8*)(Arow + ks * 32);
    af.h[1] = *(const u16x8*)(Arow + ks * 32 + 16);
#pragma unroll
    for (int nt = 0; nt < 8; ++nt) {   // 8 independent accumulators -> no RAW chain
      v16bf bfrag = *(const v16bf*)(Bs + (((ks * 8 + nt) * 32 + lane) << 4));
      acc[nt] = __builtin_amdgcn_wmma_f32_16x16x32_bf16(
          false, af.v, false, bfrag, (short)0, acc[nt], false, false);
    }
  }
  // proj = rbf @ w_rbf1 via the padded K-block 12, same C/D layout as acc
  {
    AF af;
    af.h[0] = *(const u16x8*)(Arow + 12 * 32);
    af.h[1] = *(const u16x8*)(Arow + 12 * 32 + 16);
#pragma unroll
    for (int nt = 0; nt < 8; ++nt) {
      v16bf bfrag = *(const v16bf*)(Bs + (((12 * 8 + nt) * 32 + lane) << 4));
      acc2[nt] = __builtin_amdgcn_wmma_f32_16x16x32_bf16(
          false, af.v, false, bfrag, (short)0, zero, false, false);
    }
  }

  // --- epilogue: e1 = swish(acc + b_lin); e2 = acc2 * e1 ---
#pragma unroll
  for (int nt = 0; nt < 8; ++nt) {
    int   n  = nt * 16 + m;
    float bl = WS[896 + n];
#pragma unroll
    for (int r8 = 0; r8 < 8; ++r8) {
      int  e  = wave * 16 + halfsel * 8 + r8;  // C/D layout: vgpr r8, half -> row
      long eg = base + e;
      float e1 = swishf(acc[nt][r8] + bl);
      if (eg < E) {
        out_e1[eg * HIDDEN + n] = e1;
        out_e2[eg * HIDDEN + n] = acc2[nt][r8] * e1;
      }
    }
  }
}

extern "C" void kernel_launch(void* const* d_in, const int* in_sizes, int n_in,
                              void* d_out, int out_size, void* d_ws, size_t ws_size,
                              hipStream_t stream) {
  const int*   x      = (const int*)d_in[0];
  const float* rbf    = (const float*)d_in[1];
  const int*   ei     = (const int*)d_in[2];
  const int*   ej     = (const int*)d_in[3];
  const float* emb    = (const float*)d_in[4];
  const float* w_rbf0 = (const float*)d_in[5];
  const float* b_rbf0 = (const float*)d_in[6];
  const float* w_lin  = (const float*)d_in[7];
  const float* b_lin  = (const float*)d_in[8];
  const float* w_rbf1 = (const float*)d_in[9];

  const int N = in_sizes[0];                       // atoms
  const int E = in_sizes[2];                       // edges

  unsigned short* wfrag = (unsigned short*)d_ws;                   // 106496 B
  unsigned short* hnode = (unsigned short*)((char*)d_ws + 106496); // N*128*2 B
  float* out_e1 = (float*)d_out;
  float* out_e2 = out_e1 + (size_t)E * HIDDEN;

  pack_w<<<(NFRAG + 255) / 256, 256, 0, stream>>>(w_lin, w_rbf1, wfrag);
  pack_h<<<(N * (HIDDEN / 4) + 255) / 256, 256, 0, stream>>>(x, emb, hnode, N);

  int blocks = (E + EPB - 1) / EPB;
  edge_mlp<<<blocks, THREADS, SMEM_BYTES, stream>>>(
      rbf, ei, ej, hnode, w_rbf0, b_rbf0, b_lin, wfrag,
      out_e1, out_e2, E);
}
